// OctreeDeConvBnRelu_24146306138636
// MI455X (gfx1250) — matrix operand order repaired and checked by reading, not compiled, not run on hardware
//
#include <hip/hip_runtime.h>

// Problem constants (match reference)
#define N_PTS   131072
#define C_INN   128
#define C_OUTN  128
#define KNEIGH  27
#define BN_EPS  1e-3f

typedef __attribute__((ext_vector_type(16))) __bf16 v16bf;
typedef __attribute__((ext_vector_type(8)))  float  v8f;
typedef __attribute__((ext_vector_type(4)))  float  v4f;

// Swizzled weights: per tap k, 8 ot x 4 kk x 32 lanes x 16 bf16 = 32 KB
#define WTAP_BF16  16384            // bf16 elements per tap (32 KB)
#define WTAP_V4    2048             // uint4 (16B) chunks per tap

// Workspace layout:
//  [0,512)          : per-channel sum      (128 f32)
//  [512,1024)       : per-channel sum-sq   (128 f32)
//  [1024, +884736)  : swizzled bf16 weights: [k][ot][kk][lane][16 elems]
#define WS_W_OFF_BYTES 1024

__device__ __forceinline__ __bf16 f2bf(float f) {
    unsigned u = __float_as_uint(f);
    unsigned r = (u + 0x7FFFu + ((u >> 16) & 1u)) >> 16;   // round-to-nearest-even
    return __builtin_bit_cast(__bf16, (unsigned short)r);
}

// ---------------------------------------------------------------------------
// Kernel 0: zero the accumulator (d_out) and the channel-sum scratch (b128).
// ---------------------------------------------------------------------------
__global__ void k_zero(v4f* __restrict__ out, v4f* __restrict__ sums) {
    long long i = (long long)blockIdx.x * blockDim.x + threadIdx.x;
    v4f z = {};
    if (i < (long long)N_PTS * C_OUTN / 4) out[i] = z;
    if (i < 64) sums[i] = z;
}

// ---------------------------------------------------------------------------
// Kernel W: f32 -> bf16 weight conversion, pre-swizzled into the per-lane
// B order used here, contiguous k = 16h + s as in A (32x16 tile):
//   lane (0-15): N=lane,    K = 0..15  (elems 0..15)
//   lane(16-31): N=lane-16, K = 16..31
// One thread per (k, ot, kk, lane); each writes 16 contiguous bf16 (32B).
// ---------------------------------------------------------------------------
__global__ void k_swizzle_w(const float* __restrict__ weight,
                            __bf16* __restrict__ wswz) {
    int t = blockIdx.x * blockDim.x + threadIdx.x;    // 27*8*4*32 = 27648 lanes
    if (t >= KNEIGH * 8 * 4 * 32) return;
    int lane = t & 31;
    int kk   = (t >> 5) & 3;
    int ot   = (t >> 7) & 7;
    int k    = t >> 10;
    int o    = ot * 16 + (lane & 15);
    int kbase = kk * 32 + (lane >> 4) * 16;
    __bf16* dst = wswz + (size_t)t * 16;
    const float* src = weight + ((size_t)k * C_OUTN + o) * C_INN + kbase;
#pragma unroll
    for (int e = 0; e < 16; ++e) dst[e] = f2bf(src[e]);
}

// ---------------------------------------------------------------------------
// Kernel 1: WMMA GEMM (bf16 in, f32 acc) + neighbor scatter via f32 atomics.
// Each wave owns a 16-row tile of data_in. Weights for the current tap are
// staged block-wide in LDS, double-buffered across the k-loop so the
// global->LDS copy of tap k+1 hides under the 32 WMMAs + scatter of tap k.
// The 32-step WMMA sequence is software-pipelined (depth 2) so each step's
// B fragment is already resident when its v_wmma issues.
// ---------------------------------------------------------------------------
__global__ void __launch_bounds__(256)
k_gemm_scatter(const float* __restrict__ data_in,
               const long long* __restrict__ neigh,
               const __bf16* __restrict__ wswz,
               float* __restrict__ out) {
    __shared__ uint4 lds_w[2][WTAP_V4];              // 2 x 32 KB double buffer

    const int t    = threadIdx.x;
    const int lane = t & 31;
    const int wave = t >> 5;
    const int tile = blockIdx.x * 8 + wave;          // 8192 tiles total
    const int row0 = tile * 16;
    const int m    = lane & 15;
    const int half = lane >> 4;

    // ---- Build A fragments: 16x128 f32 -> 4 x v16bf (contiguous k order, as in B) ----
    // lane half 0: elems 0..15 = K 0..15
    // lane half 1: elems 0..15 = K 16..31
    const float* arow = data_in + (size_t)(row0 + m) * C_INN;
    v16bf afrag[4];
#pragma unroll
    for (int kk = 0; kk < 4; ++kk) {
        const float* p0 = arow + kk * 32 + half * 16;       // -> elems 0..7
        const float* p1 = arow + kk * 32 + half * 16 + 8;   // -> elems 8..15
        v4f lo0 = *(const v4f*)(p0);
        v4f lo1 = *(const v4f*)(p0 + 4);
        v4f hi0 = *(const v4f*)(p1);
        v4f hi1 = *(const v4f*)(p1 + 4);
#pragma unroll
        for (int e = 0; e < 4; ++e) {
            afrag[kk][e]      = f2bf(lo0[e]);
            afrag[kk][e + 4]  = f2bf(lo1[e]);
            afrag[kk][e + 8]  = f2bf(hi0[e]);
            afrag[kk][e + 12] = f2bf(hi1[e]);
        }
    }

    const uint4* wsrc = (const uint4*)(wswz);
    float* outc = out + (lane & 15);   // + column-within-tile; row/ot added later

    // Stage tap 0 (lane-contiguous b128 -> conflict-free ds_store_b128)
#pragma unroll
    for (int i = 0; i < 8; ++i)
        lds_w[0][i * 256 + t] = wsrc[i * 256 + t];

    for (int k = 0; k < KNEIGH; ++k) {
        const int buf = k & 1;
        __syncthreads();   // staging of tap k visible; prior reads of buf^1 done

        // Kick off staging of tap k+1 into the other buffer; its global-load
        // latency overlaps the WMMA+scatter work below.
        if (k + 1 < KNEIGH) {
            const uint4* nsrc = wsrc + (size_t)(k + 1) * WTAP_V4;
#pragma unroll
            for (int i = 0; i < 8; ++i)
                lds_w[buf ^ 1][i * 256 + t] = nsrc[i * 256 + t];
            // pull tap k+2 toward L2 ahead of time (global_prefetch_b8)
            if (k + 2 < KNEIGH)
                __builtin_prefetch(wsrc + (size_t)(k + 2) * WTAP_V4 + t * 8, 0, 1);
        }

        // neighbor ids (fit in int32) for this tile's 16 rows, lanes 0..15
        int nid = -1;
        if (lane < 16) nid = (int)neigh[(size_t)(row0 + lane) * KNEIGH + k];

        // Resolve the 8 destination rows this lane will scatter:
        // C/D layout -> VGPR j holds M=j (lanes 0-15), M=j+8 (lanes 16-31)
        int dstj[8];
#pragma unroll
        for (int j = 0; j < 8; ++j)
            dstj[j] = __shfl(nid, j + (half << 3), 32);

        // 8 independent accumulator chains, 32 WMMA steps (kk-outer order),
        // software-pipelined depth 2: B fragment for step i+2 is loaded
        // before the WMMA of step i, so the inner wait is dscnt<=2, not 0.
        const v16bf* bl = (const v16bf*)(&lds_w[buf][0]) + lane;
        // fragment for linear step i (kk = i>>3, ot = i&7) at index (ot*4+kk)*32
#define FRAG(i) bl[(((i) & 7) * 4 + ((i) >> 3)) * 32]
        v8f acc[8];
#pragma unroll
        for (int ot = 0; ot < 8; ++ot) acc[ot] = (v8f){};
        v16bf b0 = FRAG(0);
        v16bf b1 = FRAG(1);
#pragma unroll
        for (int i = 0; i < 32; ++i) {
            v16bf bcur = (i & 1) ? b1 : b0;
            if (i + 2 < 32) {
                if (i & 1) b1 = FRAG(i + 2); else b0 = FRAG(i + 2);
            }
            const int kk = i >> 3, ot = i & 7;
            acc[ot] = __builtin_amdgcn_wmma_f32_16x16x32_bf16(
                false, afrag[kk], false, bcur, (short)0, acc[ot], false, false);
        }
#undef FRAG

        // Scatter, j-outer: one branch + one address per destination row,
        // 8 column tiles at immediate offsets 0,64,...,448 bytes.
#pragma unroll
        for (int j = 0; j < 8; ++j) {
            int dst = dstj[j];
            if (dst >= 0) {
                float* p = outc + ((unsigned)dst << 7);   // dst*128 fits 32 bits
#pragma unroll
                for (int ot = 0; ot < 8; ++ot)
                    atomicAdd(p + ot * 16, acc[ot][j]);
            }
        }
    }
}

// ---------------------------------------------------------------------------
// Kernel 2: per-channel sum / sum-of-squares over the accumulator (b128 reads).
// ---------------------------------------------------------------------------
__global__ void k_reduce(const float* __restrict__ out,
                         float* __restrict__ sum, float* __restrict__ sumsq) {
    int c4 = (threadIdx.x & 31) * 4;       // 4 consecutive channels
    int rg = threadIdx.x >> 5;             // 8 row-groups per block
    v4f s = {}, q = {};
    for (int row = blockIdx.x * 8 + rg; row < N_PTS; row += gridDim.x * 8) {
        v4f v = *(const v4f*)(out + (size_t)row * C_OUTN + c4);
#pragma unroll
        for (int e = 0; e < 4; ++e) { s[e] += v[e]; q[e] += v[e] * v[e]; }
    }
#pragma unroll
    for (int e = 0; e < 4; ++e) {
        atomicAdd(sum + c4 + e, s[e]);
        atomicAdd(sumsq + c4 + e, q[e]);
    }
}

// ---------------------------------------------------------------------------
// Kernel 3: in-place BatchNorm + ReLU (b128 read-modify-write).
// ---------------------------------------------------------------------------
__global__ void k_bn_relu(float* __restrict__ out,
                          const float* __restrict__ sum,
                          const float* __restrict__ sumsq,
                          const float* __restrict__ gamma,
                          const float* __restrict__ beta) {
    long long i = (long long)blockIdx.x * blockDim.x + threadIdx.x;
    if (i >= (long long)N_PTS * C_OUTN / 4) return;
    int c4 = (int)((i * 4) & 127);
    const float invN = 1.0f / (float)N_PTS;
    v4f v = *((v4f*)out + i);
#pragma unroll
    for (int e = 0; e < 4; ++e) {
        int c = c4 + e;
        float mean = sum[c] * invN;
        float var  = sumsq[c] * invN - mean * mean;
        float y = (v[e] - mean) * __frsqrt_rn(var + BN_EPS) * gamma[c] + beta[c];
        v[e] = y > 0.0f ? y : 0.0f;
    }
    *((v4f*)out + i) = v;
}

// ---------------------------------------------------------------------------
extern "C" void kernel_launch(void* const* d_in, const int* in_sizes, int n_in,
                              void* d_out, int out_size, void* d_ws, size_t ws_size,
                              hipStream_t stream) {
    const float*     data_in = (const float*)d_in[0];
    const long long* neigh   = (const long long*)d_in[1];   // int64 neighbor ids
    const float*     weight  = (const float*)d_in[2];
    const float*     gamma   = (const float*)d_in[3];
    const float*     beta    = (const float*)d_in[4];

    float*  out   = (float*)d_out;
    float*  sum   = (float*)d_ws;            // 128 f32
    float*  sumsq = sum + 128;               // 128 f32
    __bf16* wswz  = (__bf16*)((char*)d_ws + WS_W_OFF_BYTES);

    const int vec_total = (int)((long long)N_PTS * C_OUTN / 4);   // 4,194,304
    const int vb = (vec_total + 255) / 256;                       // 16,384 blocks

    k_zero<<<vb, 256, 0, stream>>>((v4f*)out, (v4f*)sum);
    k_swizzle_w<<<(KNEIGH * 8 * 4 * 32 + 255) / 256, 256, 0, stream>>>(weight, wswz);
    k_gemm_scatter<<<N_PTS / 16 / 8, 256, 0, stream>>>(data_in, neigh, wswz, out);
    k_reduce<<<256, 256, 0, stream>>>(out, sum, sumsq);
    k_bn_relu<<<vb, 256, 0, stream>>>(out, sum, sumsq, gamma, beta);
}
